// AttenuateDenoiser_87763361727190
// MI455X (gfx1250) — compile-verified, matrix operand order checked by reading, not run on hardware
//
#include <hip/hip_runtime.h>
#include <hip/hip_bf16.h>
#include <stdint.h>

typedef __attribute__((ext_vector_type(16))) __bf16 v16bf;
typedef __attribute__((ext_vector_type(8)))  __bf16 v8bf;
typedef __attribute__((ext_vector_type(8)))  float  v8f;

#define BATCH   8
#define NMODES  256
#define SEG     4096
#define L0TOT   160256
#define LOUT    160000

// ----------------------------- WMMA GEMM -----------------------------------
// D[m,n] = sum_k A[m,k] * B[k,n], bf16 inputs, f32 accumulate.
// Guard-free staging (A K-zero-padded, B has allocation slack), double-buffered
// LDS with one barrier per K-tile, async global->LDS DMA for the A tile.
#define BM 64
#define BN 32
#define BK 32
#define ASTR 48
#define BSTR 48

enum { BMODE_PLAIN = 0, BMODE_DOWNMIX = 1 };

struct GemmArgs {
  const __bf16* A; int lda;                       // Mpad x Kpad, K zero-padded
  const __bf16* B; int b_batch; int ldb;          // per-batch source
  int rsh; int rmask;                             // r = 1<<rsh (DOWNMIX)
  float*  Df; int df_batch; int ldd; int dstride; // f32 out (col stride)
  __bf16* Dh; int dh_batch; int ldh;              // bf16 out (col stride 1)
  int M, N, Kpad;
};

// gfx1250 async DMA: global -> LDS, 16B per lane, tracked by ASYNCcnt.
__device__ __forceinline__ void async_copy_b128(unsigned lds_addr, const __bf16* gaddr) {
  asm volatile("global_load_async_to_lds_b128 %0, %1, off"
               :: "v"(lds_addr), "v"(gaddr) : "memory");
}
__device__ __forceinline__ void wait_async() {
  asm volatile("s_wait_asynccnt 0x0" ::: "memory");
}

template<int MODE, bool SF32, bool SBF16>
__global__ __launch_bounds__(256) void gemm_bf16_kernel(GemmArgs g) {
  __shared__ __align__(16) __bf16 a_s[2][BM * ASTR];
  __shared__ __align__(16) __bf16 b_s[2][BN * BSTR];  // transposed: [n][k]
  const int b    = blockIdx.z;
  const int m0   = blockIdx.y * BM;
  const int n0   = blockIdx.x * BN;
  const int tid  = threadIdx.x;
  const int wave = tid >> 5;
  const int lane = tid & 31;
  const int mt   = wave & 3;                        // 4 m-tiles
  const int nt   = wave >> 2;                       // 2 n-tiles

  // staging maps (exact cover of tiles by 256 threads)
  const __bf16* __restrict__ Asrc = g.A + (m0 + (tid >> 2)) * g.lda + (tid & 3) * 8;
  const __bf16* __restrict__ Bsrc = g.B + b * g.b_batch;
  const int bk = tid >> 3, bn = (tid & 7) * 4;
  const int a_st_off = (tid >> 2) * ASTR + (tid & 3) * 8;
  const int b_st_off = bn * BSTR + bk;

  // fragment offsets (fixed per lane)
  const int khalf  = lane >> 4;
  const int a_f_off = (mt * 16 + (lane & 15)) * ASTR + khalf * 8;
  const int b_f_off = (nt * 16 + (lane & 15)) * BSTR + khalf * 8;

  auto stage = [&](int k0, int p) {
    // A tile: async DMA straight into LDS (no VGPR round-trip)
    async_copy_b128((unsigned)(uintptr_t)&a_s[p][a_st_off], Asrc + k0);
    // B tile: coalesced loads + transpose-scatter into LDS
    __bf16* bst = &b_s[p][b_st_off];
    if (MODE == BMODE_PLAIN) {
      int roff = (k0 + bk) * g.ldb + n0 + bn;
#pragma unroll
      for (int i = 0; i < 4; ++i) bst[i * BSTR] = Bsrc[roff + i];
      if (k0 + BK < g.Kpad)
        __builtin_prefetch(Bsrc + roff + BK * g.ldb, 0, 0);  // global_prefetch_b8
    } else {
      int gk = k0 + bk;
      int base = (gk >> g.rsh) * g.ldb + (gk & g.rmask);
#pragma unroll
      for (int i = 0; i < 4; ++i) bst[i * BSTR] = Bsrc[base + ((n0 + bn + i) << g.rsh)];
    }
  };

  stage(0, 0);
  v8f acc = {};
  int p = 0;
  for (int k0 = 0; k0 < g.Kpad; k0 += BK, p ^= 1) {
    wait_async();          // own async DMA done
    __syncthreads();       // buffer p published by all waves
    if (k0 + BK < g.Kpad) stage(k0 + BK, p ^ 1);   // overlap next tile with WMMA
    // ---- fragments (documented 16-bit A/B VGPR layout) ----
    const __bf16* a_f0 = &a_s[p][a_f_off];
    const __bf16* b_f0 = &b_s[p][b_f_off];
    v8bf alo = *(const v8bf*)(a_f0);
    v8bf ahi = *(const v8bf*)(a_f0 + 16);
    v8bf blo = *(const v8bf*)(b_f0);
    v8bf bhi = *(const v8bf*)(b_f0 + 16);
    v16bf afrag, bfrag;
#pragma unroll
    for (int i = 0; i < 8; ++i) {
      afrag[i] = alo[i]; afrag[i + 8] = ahi[i];
      bfrag[i] = blo[i]; bfrag[i + 8] = bhi[i];
    }
    acc = __builtin_amdgcn_wmma_f32_16x16x32_bf16(
        false, afrag, false, bfrag, (short)0, acc, false, false);
  }
  // ---- writeback (col = lane%16, rows mg0+j; mask only here) ----
  const int ng  = n0 + nt * 16 + (lane & 15);
  const int mg0 = m0 + mt * 16 + ((lane >> 4) << 3);
  if (ng < g.N) {
    float*  dp = SF32  ? (g.Df + b * g.df_batch + mg0 * g.ldd + ng * g.dstride) : (float*)0;
    __bf16* hp = SBF16 ? (g.Dh + b * g.dh_batch + mg0 * g.ldh + ng) : (__bf16*)0;
#pragma unroll
    for (int j = 0; j < 8; ++j) {
      if (mg0 + j < g.M) {
        if (SF32)  dp[j * g.ldd] = acc[j];
        if (SBF16) hp[j * g.ldh] = (__bf16)acc[j];
      }
    }
  }
}

// ------------------------- SSM diagonal complex scan ------------------------
// s <- lambda * s + z ; y = Re(s).  One thread per (batch, mode): 2048 scans.
__global__ void ssm_scan_kernel(const __bf16* __restrict__ z,
                                __bf16* __restrict__ y,
                                float2* __restrict__ state,
                                const float2* __restrict__ lambda, int segLen) {
  int id = blockIdx.x * blockDim.x + threadIdx.x;   // b*256 + n
  int n  = id & (NMODES - 1);
  float2 lam = lambda[n];
  float2 s   = state[id];
  const __bf16* zp = z + (long)id * segLen;
  __bf16*       yp = y + (long)id * segLen;
  if ((segLen & 7) == 0) {                          // b128 I/O path
    const v8bf* zv = (const v8bf*)zp;
    v8bf*       yv = (v8bf*)yp;
    for (int v = 0; v < (segLen >> 3); ++v) {
      v8bf zz = zv[v];
      v8bf yy;
#pragma unroll
      for (int j = 0; j < 8; ++j) {
        float zvl = (float)zz[j];
        float re = lam.x * s.x - lam.y * s.y + zvl;
        float im = lam.x * s.y + lam.y * s.x;
        s.x = re; s.y = im;
        yy[j] = (__bf16)re;
      }
      yv[v] = yy;
    }
  } else {
    for (int l = 0; l < segLen; ++l) {
      float zvl = (float)zp[l];
      float re = lam.x * s.x - lam.y * s.y + zvl;
      float im = lam.x * s.y + lam.y * s.x;
      s.x = re; s.y = im;
      yp[l] = (__bf16)re;
    }
  }
  state[id] = s;
}

__global__ void zero_states_kernel(float2* state) {
  int id = blockIdx.x * blockDim.x + threadIdx.x;
  state[id] = make_float2(0.f, 0.f);
}

// lambda_n = exp(-dt*softplus(A0) + i*dt*A1); B_hat = B * dt (bf16, K padded)
__global__ void prep_modes_kernel(const float* __restrict__ A,
                                  const float* __restrict__ B,
                                  const float* __restrict__ log_dt,
                                  float2* __restrict__ lambda,
                                  __bf16* __restrict__ Bhat, int ch, int kpad) {
  int n = blockIdx.x * blockDim.x + threadIdx.x;
  if (n >= NMODES) return;
  float dt = __expf(log_dt[n]);
  float a0 = A[2 * n], a1 = A[2 * n + 1];
  float sp = (a0 > 20.f) ? a0 : __logf(1.f + __expf(a0));
  float mag = __expf(-dt * sp);
  float om  = dt * a1;
  lambda[n] = make_float2(mag * __cosf(om), mag * __sinf(om));
  for (int c = 0; c < kpad; ++c)
    Bhat[n * kpad + c] = (c < ch) ? (__bf16)(B[n * ch + c] * dt) : (__bf16)0.0f;
}

// ----------------------------- elementwise ----------------------------------
__global__ void to_bf16_kernel(const float* __restrict__ x, __bf16* __restrict__ y, int n) {
  int i = blockIdx.x * blockDim.x + threadIdx.x;
  if (i < n) y[i] = (__bf16)x[i];
}

__global__ void dwconv_kernel(const float* __restrict__ x, const float* __restrict__ w,
                              const float* __restrict__ bias,
                              float* __restrict__ y, __bf16* __restrict__ ybf,
                              int ch, int L, int total) {
  int i = blockIdx.x * blockDim.x + threadIdx.x;
  if (i >= total) return;
  int l = i % L;
  int c = (i / L) % ch;
  float xm = (l > 0)     ? x[i - 1] : 0.f;
  float x0 = x[i];
  float xp = (l < L - 1) ? x[i + 1] : 0.f;
  float v = w[c * 3] * xm + w[c * 3 + 1] * x0 + w[c * 3 + 2] * xp + bias[c];
  y[i] = v;
  ybf[i] = (__bf16)v;
}

__global__ void ln_silu_kernel(const float* __restrict__ x, const float* __restrict__ gam,
                               const float* __restrict__ bet,
                               float* __restrict__ y, __bf16* __restrict__ ybf,
                               int ch, int L, int cols) {
  int col = blockIdx.x * blockDim.x + threadIdx.x;    // over BATCH*L
  if (col >= cols) return;
  int b = col / L, l = col % L;
  const float* xp = x + (long)b * ch * L + l;
  float mu = 0.f, s2 = 0.f;
  for (int c = 0; c < ch; ++c) { float v = xp[(long)c * L]; mu += v; s2 += v * v; }
  mu /= (float)ch;
  float var = s2 / (float)ch - mu * mu;
  float inv = __frsqrt_rn(var + 1e-5f);
  for (int c = 0; c < ch; ++c) {
    float v = (xp[(long)c * L] - mu) * inv * gam[c] + bet[c];
    float o = v / (1.f + __expf(-v));
    long oi = (long)b * ch * L + (long)c * L + l;
    y[oi] = o;
    ybf[oi] = (__bf16)o;
  }
}

__global__ void silu_kernel(float* __restrict__ x, __bf16* __restrict__ ybf, int n) {
  int i = blockIdx.x * blockDim.x + threadIdx.x;
  if (i >= n) return;
  float v = x[i];
  float o = v / (1.f + __expf(-v));
  x[i] = o;
  ybf[i] = (__bf16)o;
}

__global__ void add_kernel(const float* __restrict__ a, const float* __restrict__ b,
                           float* __restrict__ y, __bf16* __restrict__ ybf, int n) {
  int i = blockIdx.x * blockDim.x + threadIdx.x;
  if (i >= n) return;
  float v = a[i] + b[i];
  y[i] = v;
  ybf[i] = (__bf16)v;
}

__global__ void pad_audio_kernel(const float* __restrict__ audio,
                                 float* __restrict__ xf, __bf16* __restrict__ xh, int total) {
  int i = blockIdx.x * blockDim.x + threadIdx.x;
  if (i >= total) return;
  int b = i / L0TOT, t = i % L0TOT;
  float v = (t < LOUT) ? audio[(long)b * LOUT + t] : 0.f;
  xf[i] = v;
  xh[i] = (__bf16)v;
}

__global__ void copy_out_kernel(const float* __restrict__ x, float* __restrict__ out, int total) {
  int i = blockIdx.x * blockDim.x + threadIdx.x;
  if (i >= total) return;
  int b = i / LOUT, t = i % LOUT;
  out[i] = x[(long)b * L0TOT + t];
}

// pack down-mix W(c_in,c_out,r) -> A[d][k], k = c*r+rr, K zero-padded to kpad
__global__ void pack_down_w_kernel(const float* __restrict__ W, __bf16* __restrict__ A,
                                   int cin, int cout, int r, int kpad, int total) {
  int i = blockIdx.x * blockDim.x + threadIdx.x;
  if (i >= total) return;
  int d = i / kpad, k = i % kpad;
  int kk = cin * r;
  int c = k / r, rr = k % r;
  A[i] = (k < kk) ? (__bf16)W[((long)c * cout + d) * r + rr] : (__bf16)0.0f;
}

// pack up-mix W(c_in,c_out,r) -> A[rr][d][c], c zero-padded to kpad
__global__ void pack_up_w_kernel(const float* __restrict__ W, __bf16* __restrict__ A,
                                 int cin, int cout, int r, int kpad, int total) {
  int i = blockIdx.x * blockDim.x + threadIdx.x;
  if (i >= total) return;
  int per = cout * kpad;
  int rr = i / per, rem = i % per, d = rem / kpad, c = rem % kpad;
  A[i] = (c < cin) ? (__bf16)W[((long)c * cout + d) * r + rr] : (__bf16)0.0f;
}

// =============================== host side ==================================
static inline int cdiv(int a, int b) { return (a + b - 1) / b; }
static inline int ceilmul(int a, int m) { return cdiv(a, m) * m; }
static inline int log2r(int r) { return r == 4 ? 2 : 1; }

struct BlockP { const float *A, *B, *C, *conv_b, *conv_w, *ln_b, *ln_g, *log_dt; };

// jax tree_flatten order: dict keys sorted alphabetically.
// ch>1 block: A,B,C,conv_b,conv_w,ln_b,ln_g,log_dt ; ch==1: A,B,C,log_dt
static BlockP read_block(void* const* d_in, int& i, int ch) {
  BlockP p{};
  p.A = (const float*)d_in[i++];
  p.B = (const float*)d_in[i++];
  p.C = (const float*)d_in[i++];
  if (ch > 1) {
    p.conv_b = (const float*)d_in[i++];
    p.conv_w = (const float*)d_in[i++];
    p.ln_b   = (const float*)d_in[i++];
    p.ln_g   = (const float*)d_in[i++];
  }
  p.log_dt = (const float*)d_in[i++];
  return p;
}

struct Scratch {
  float2* lambda; float2* states;
  __bf16 *Abh, *Ac, *Wpack, *zseg, *yseg;
};

static void run_block(hipStream_t st, const BlockP& p, int ch, int L, bool act,
                      const float* x_f, const __bf16* x_bf,
                      float* tmp_f, __bf16* tmp_bf,
                      float* out_f, __bf16* out_bf, const Scratch& sc) {
  int kpad = ceilmul(ch, 32);
  prep_modes_kernel<<<1, 256, 0, st>>>(p.A, p.B, p.log_dt, sc.lambda, sc.Abh, ch, kpad);
  to_bf16_kernel<<<cdiv(ch * NMODES, 256), 256, 0, st>>>(p.C, sc.Ac, ch * NMODES);

  const __bf16* xc_bf = x_bf;
  if (ch > 1) {
    int n = BATCH * ch * L;
    dwconv_kernel<<<cdiv(n, 256), 256, 0, st>>>(x_f, p.conv_w, p.conv_b, tmp_f, tmp_bf, ch, L, n);
    xc_bf = tmp_bf;
  }
  zero_states_kernel<<<BATCH, 256, 0, st>>>(sc.states);

  for (int s0 = 0; s0 < L; s0 += SEG) {
    int sl = (L - s0 < SEG) ? (L - s0) : SEG;
    GemmArgs gz{};
    gz.A = sc.Abh; gz.lda = kpad;
    gz.B = xc_bf + s0; gz.b_batch = ch * L; gz.ldb = L;
    gz.Dh = sc.zseg; gz.dh_batch = NMODES * sl; gz.ldh = sl;
    gz.M = NMODES; gz.N = sl; gz.Kpad = kpad;
    gemm_bf16_kernel<BMODE_PLAIN, false, true>
        <<<dim3(cdiv(sl, BN), cdiv(NMODES, BM), BATCH), 256, 0, st>>>(gz);

    ssm_scan_kernel<<<BATCH, 256, 0, st>>>(sc.zseg, sc.yseg, sc.states, sc.lambda, sl);

    GemmArgs go{};
    go.A = sc.Ac; go.lda = NMODES;
    go.B = sc.yseg; go.b_batch = NMODES * sl; go.ldb = sl;
    go.Df = out_f + s0; go.df_batch = ch * L; go.ldd = L; go.dstride = 1;
    go.M = ch; go.N = sl; go.Kpad = NMODES;
    gemm_bf16_kernel<BMODE_PLAIN, true, false>
        <<<dim3(cdiv(sl, BN), cdiv(ch, BM), BATCH), 256, 0, st>>>(go);
  }

  if (act) {
    if (ch > 1)
      ln_silu_kernel<<<cdiv(BATCH * L, 256), 256, 0, st>>>(out_f, p.ln_g, p.ln_b,
                                                           out_f, out_bf, ch, L, BATCH * L);
    else
      silu_kernel<<<cdiv(BATCH * L, 256), 256, 0, st>>>(out_f, out_bf, BATCH * L);
  }
}

extern "C" void kernel_launch(void* const* d_in, const int* in_sizes, int n_in,
                              void* d_out, int out_size, void* d_ws, size_t ws_size,
                              hipStream_t stream) {
  (void)in_sizes; (void)n_in; (void)out_size; (void)ws_size;
  static const int CH[7]  = {1, 16, 32, 64, 96, 128, 256};
  static const int RES[6] = {4, 4, 2, 2, 2, 2};
  int L[7]; L[0] = L0TOT;
  for (int i = 0; i < 6; ++i) L[i + 1] = L[i] / RES[i];

  const float* audio = (const float*)d_in[0];
  int pidx = 1;
  // params flattened (sorted keys): down, hid, last, up
  BlockP down_blk[6]; const float* down_mixw[6];
  for (int i = 0; i < 6; ++i) { down_blk[i] = read_block(d_in, pidx, CH[i]); down_mixw[i] = (const float*)d_in[pidx++]; }
  BlockP hid_blk[2];
  for (int j = 0; j < 2; ++j) hid_blk[j] = read_block(d_in, pidx, 256);
  BlockP last_blk[2];
  last_blk[0] = read_block(d_in, pidx, 1);
  last_blk[1] = read_block(d_in, pidx, 1);
  BlockP up_blk[6]; const float* up_mixw[6];
  for (int i = 0; i < 6; ++i) { up_blk[i] = read_block(d_in, pidx, CH[i]); up_mixw[i] = (const float*)d_in[pidx++]; }

  // ---- workspace bump allocator (deterministic each call; +256B slack) ----
  char* wp = (char*)d_ws;
  auto alloc = [&](size_t bytes) -> void* {
    uintptr_t p = ((uintptr_t)wp + 255) & ~(uintptr_t)255;
    wp = (char*)(p + bytes + 256);
    return (void*)p;
  };
  const size_t MAXE  = (size_t)BATCH * 16 * 40064;  // largest activation (level 1)
  const size_t SLACK = 31u * L0TOT + 64;            // K-overread slack for GEMM-B
  float*  fb[3]; __bf16* hb[3];
  for (int i = 0; i < 3; ++i) {
    fb[i] = (float*)alloc(MAXE * sizeof(float));
    hb[i] = (__bf16*)alloc((MAXE + SLACK) * sizeof(__bf16));
  }
  float* skip[6];
  for (int i = 0; i < 6; ++i) skip[i] = (float*)alloc((size_t)BATCH * CH[i] * L[i] * sizeof(float));
  Scratch sc;
  sc.lambda = (float2*)alloc(NMODES * sizeof(float2));
  sc.states = (float2*)alloc((size_t)BATCH * NMODES * sizeof(float2));
  sc.Abh    = (__bf16*)alloc((size_t)NMODES * 256 * sizeof(__bf16));
  sc.Ac     = (__bf16*)alloc((size_t)256 * 256 * sizeof(__bf16));
  sc.Wpack  = (__bf16*)alloc((size_t)131072 * sizeof(__bf16));
  sc.zseg   = (__bf16*)alloc((size_t)BATCH * NMODES * SEG * sizeof(__bf16));
  sc.yseg   = (__bf16*)alloc((size_t)BATCH * NMODES * SEG * sizeof(__bf16));

  // ---- input ----
  float* cur_f = fb[0]; __bf16* cur_bf = hb[0]; int fi = 0;
  pad_audio_kernel<<<cdiv(BATCH * L0TOT, 256), 256, 0, stream>>>(audio, cur_f, cur_bf, BATCH * L0TOT);

  // ---- down path ----
  for (int i = 0; i < 6; ++i) {
    hipMemcpyAsync(skip[i], cur_f, (size_t)BATCH * CH[i] * L[i] * sizeof(float),
                   hipMemcpyDeviceToDevice, stream);
    int o = (fi + 1) % 3, t = (fi + 2) % 3;
    run_block(stream, down_blk[i], CH[i], L[i], true,
              cur_f, cur_bf, fb[t], hb[t], fb[o], hb[o], sc);
    int r = RES[i], cin = CH[i], cout = CH[i + 1], T = L[i + 1];
    int kpad = ceilmul(cin * r, 32);
    int wn = cout * kpad;
    pack_down_w_kernel<<<cdiv(wn, 256), 256, 0, stream>>>(down_mixw[i], sc.Wpack, cin, cout, r, kpad, wn);
    GemmArgs gm{};
    gm.A = sc.Wpack; gm.lda = kpad;
    gm.B = hb[o]; gm.b_batch = cin * L[i]; gm.ldb = L[i];
    gm.rsh = log2r(r); gm.rmask = r - 1;
    gm.Df = fb[t]; gm.df_batch = cout * T; gm.ldd = T; gm.dstride = 1;
    gm.M = cout; gm.N = T; gm.Kpad = kpad;
    gemm_bf16_kernel<BMODE_DOWNMIX, true, false>
        <<<dim3(cdiv(T, BN), cdiv(cout, BM), BATCH), 256, 0, stream>>>(gm);
    cur_f = fb[t]; cur_bf = hb[t]; fi = t;   // bf16 not needed (conv re-emits)
  }

  // ---- hidden blocks (ch=256, L=626) ----
  for (int j = 0; j < 2; ++j) {
    int o = (fi + 1) % 3, t = (fi + 2) % 3;
    run_block(stream, hid_blk[j], 256, L[6], true,
              cur_f, cur_bf, fb[t], hb[t], fb[o], hb[o], sc);
    cur_f = fb[o]; cur_bf = hb[o]; fi = o;
  }

  // ---- up path (reversed) ----
  for (int i = 5; i >= 0; --i) {
    int r = RES[i], cin = CH[i + 1], cout = CH[i], T = L[i + 1], Lout = L[i];
    int kpad = ceilmul(cin, 32);
    int wn = r * cout * kpad;
    pack_up_w_kernel<<<cdiv(wn, 256), 256, 0, stream>>>(up_mixw[i], sc.Wpack, cin, cout, r, kpad, wn);
    int o = (fi + 1) % 3, t = (fi + 2) % 3;
    for (int rr = 0; rr < r; ++rr) {
      GemmArgs gu{};
      gu.A = sc.Wpack + rr * cout * kpad; gu.lda = kpad;
      gu.B = cur_bf; gu.b_batch = cin * T; gu.ldb = T;
      gu.Df = fb[o] + rr; gu.df_batch = cout * Lout; gu.ldd = Lout; gu.dstride = r;
      gu.M = cout; gu.N = T; gu.Kpad = kpad;
      gemm_bf16_kernel<BMODE_PLAIN, true, false>
          <<<dim3(cdiv(T, BN), cdiv(cout, BM), BATCH), 256, 0, stream>>>(gu);
    }
    int n = BATCH * cout * Lout;
    add_kernel<<<cdiv(n, 256), 256, 0, stream>>>(fb[o], skip[i], fb[t], hb[t], n);
    run_block(stream, up_blk[i], cout, Lout, true,
              fb[t], hb[t], fb[fi], hb[fi], fb[o], hb[o], sc);
    cur_f = fb[o]; cur_bf = hb[o]; fi = o;
  }

  // ---- final ch=1 blocks ----
  {
    int o = (fi + 1) % 3, t = (fi + 2) % 3;
    run_block(stream, last_blk[0], 1, L0TOT, true,
              cur_f, cur_bf, fb[t], hb[t], fb[o], hb[o], sc);
    cur_f = fb[o]; cur_bf = hb[o]; fi = o;
  }
  {
    int o = (fi + 1) % 3, t = (fi + 2) % 3;
    run_block(stream, last_blk[1], 1, L0TOT, false,
              cur_f, cur_bf, fb[t], hb[t], fb[o], hb[o], sc);
    cur_f = fb[o]; fi = o;
  }

  copy_out_kernel<<<cdiv(BATCH * LOUT, 256), 256, 0, stream>>>(cur_f, (float*)d_out, BATCH * LOUT);
}
